// GNNML3Layer_84086869721473
// MI455X (gfx1250) — matrix operand order repaired and checked by reading, not compiled, not run on hardware
//
#include <hip/hip_runtime.h>
#include <hip/hip_bf16.h>

typedef _Float16 v16h __attribute__((ext_vector_type(16)));
typedef _Float16 v8h  __attribute__((ext_vector_type(8)));
typedef float    v8f  __attribute__((ext_vector_type(8)));
typedef unsigned int v4u __attribute__((ext_vector_type(4)));
typedef int v4i __attribute__((ext_vector_type(4)));
typedef int v8i __attribute__((ext_vector_type(8)));

static __device__ __forceinline__ v8f wmma16(v16h a, v16h b, v8f c) {
  // D = A(16x32 f16) * B(32x16 f16) + C(16x16 f32)
  return __builtin_amdgcn_wmma_f32_16x16x32_f16(
      /*neg_a=*/false, a, /*neg_b=*/false, b,
      /*c_mod=*/(short)0, c, /*reuse_a=*/false, /*reuse_b=*/false);
}

static __device__ __forceinline__ v16h cat8(v8h lo, v8h hi) {
  return __builtin_shufflevector(lo, hi, 0, 1, 2, 3, 4, 5, 6, 7, 8, 9, 10, 11,
                                 12, 13, 14, 15);
}
static __device__ __forceinline__ v8h ldv8h(const _Float16* p) {
  return *(const v8h*)p;
}

static __device__ __forceinline__ float fast_rcp(float x) {
#if __has_builtin(__builtin_amdgcn_rcpf)
  return __builtin_amdgcn_rcpf(x);  // single v_rcp_f32
#else
  return 1.0f / x;
#endif
}
// Branch-free tanh: copysign(1 - 2*rcp(exp(2|x|)+1), x). v_exp + v_rcp only.
static __device__ __forceinline__ float fast_tanh(float v) {
  float e2 = __expf(2.0f * fabsf(v));
  float r = 1.0f - 2.0f * fast_rcp(e2 + 1.0f);
  return copysignf(r, v);
}

// A fragment (16x32 f16 layout) built from a row of f32 data; the two 8-elem
// runs per lane are contiguous -> compiler merges into b128 loads.
static __device__ __forceinline__ v16h afrag_f32(const float* rowp, int kb,
                                                 int hi) {
  const float* p0 = rowp + kb + hi * 16;
  const float* p1 = rowp + kb + hi * 16 + 8;
  v16h a;
#pragma unroll
  for (int j = 0; j < 8; ++j) {
    a[j] = (_Float16)p0[j];
    a[j + 8] = (_Float16)p1[j];
  }
  return a;
}

// ---------------------------------------------------------------------------
// TDM: 1-D tensor load (f32 elements) global -> LDS via Tensor Data Mover.
// D# per CDNA5 ISA; tensor_dim0 < tile_dim0 zero-fills.
// ---------------------------------------------------------------------------
#if __has_builtin(__builtin_amdgcn_tensor_load_to_lds)
#define HAVE_TDM 1
static __device__ __forceinline__ void tdm_load_1d_f32(
    void* lds_ptr, const void* gptr, unsigned tile_elems,
    unsigned tensor_elems) {
  unsigned lds_off = (unsigned)(unsigned long long)lds_ptr;  // LDS byte offset
  unsigned long long ga = (unsigned long long)gptr;
  v4u g0;
  g0[0] = 1u;                        // count=1, user descriptor
  g0[1] = lds_off;                   // lds_addr
  g0[2] = (unsigned)ga;              // global_addr[31:0]
  g0[3] = (unsigned)((ga >> 32) & 0x01FFFFFFu) | 0x80000000u;  // [56:32]|type=2
  v8i g1;
  g1[0] = (int)(2u << 16);                           // data_size = 4 bytes
  g1[1] = (int)((tensor_elems & 0xFFFFu) << 16);     // tensor_dim0[15:0]
  g1[2] = (int)((tensor_elems >> 16) | (1u << 16));  // td0[31:16], td1=1
  g1[3] = (int)(tile_elems << 16);                   // tile_dim0
  g1[4] = 0;                                         // tile_dim1/2 unused
  g1[5] = (int)tensor_elems;                         // tensor_dim0_stride
  g1[6] = 0;
  g1[7] = 0;
  v4i gz = {0, 0, 0, 0};
#if __clang_major__ >= 23
  v8i gz8 = {0, 0, 0, 0, 0, 0, 0, 0};
  __builtin_amdgcn_tensor_load_to_lds(g0, g1, gz, gz, gz8, 0);
#else
  __builtin_amdgcn_tensor_load_to_lds(g0, g1, gz, gz, 0);
#endif
}
#else
#define HAVE_TDM 0
#endif

// ---------------------------------------------------------------------------
// Kernel 0: zero the spectral accumulator [N,128]
// ---------------------------------------------------------------------------
__global__ void zero_kernel(float4* __restrict__ p, long long n4) {
  long long i = (long long)blockIdx.x * blockDim.x + threadIdx.x;
  if (i < n4) p[i] = make_float4(0.f, 0.f, 0.f, 0.f);
}

// ---------------------------------------------------------------------------
// Kernel 1: edge feature MLP -> e[E,8]
// Block = 128 thr (4 waves); wave owns a 16-edge tile; the block's 64x32 f32
// edge_attr tile (8KB contiguous) is DMA'd into LDS by the TDM. Weight panels
// stored transposed+padded so B fragments are contiguous ds_load_b128 pairs.
// ---------------------------------------------------------------------------
__global__ __launch_bounds__(128) void edge_mlp_kernel(
    const float* __restrict__ ea,
    const float* __restrict__ W1, const float* __restrict__ W2,
    const float* __restrict__ W3, const float* __restrict__ W4,
    float* __restrict__ e_out, int E) {
  __shared__ __attribute__((aligned(32))) float eas[64 * 32];        // 8KB
  __shared__ __attribute__((aligned(32))) _Float16 w1sT[64 * 40];    // [n][k]
  __shared__ __attribute__((aligned(32))) _Float16 w2sT[64 * 40];
  __shared__ __attribute__((aligned(32))) _Float16 w3sT[64 * 40];
  __shared__ __attribute__((aligned(32))) _Float16 w4sT[16 * 136];   // [n][k]
  __shared__ __attribute__((aligned(32))) _Float16 stage[4][16 * 128];

  int tid = threadIdx.x;
  int wave = tid >> 5, lane = tid & 31;
  int row = lane & 15, hi = lane >> 4;
  long long ebase0 = (long long)blockIdx.x * 64;
  long long ebase = ebase0 + wave * 16;

#if HAVE_TDM
  if (wave == 0) {
    long long remain = (long long)E * 32 - ebase0 * 32;
    unsigned telems =
        (remain >= 2048) ? 2048u : (unsigned)(remain > 0 ? remain : 0);
    tdm_load_1d_f32(eas, ea + ebase0 * 32, 2048u, telems);
    __builtin_amdgcn_s_wait_tensorcnt(0);
  }
#else
  for (int i = tid; i < 2048; i += 128) {
    long long idx = ebase0 * 32 + i;
    eas[i] = (idx < (long long)E * 32) ? ea[idx] : 0.f;
  }
#endif

  // transpose weights into LDS: wT[n*pitch + k]
  for (int i = tid; i < 2048; i += 128) {
    int k = i >> 6, n = i & 63;  // W[32][64]
    w1sT[n * 40 + k] = (_Float16)W1[i];
    w2sT[n * 40 + k] = (_Float16)W2[i];
    w3sT[n * 40 + k] = (_Float16)W3[i];
  }
  for (int i = tid; i < 2048; i += 128) {
    int k = i >> 4, n = i & 15;  // [128][16] padded from W4 [128][8]
    w4sT[n * 136 + k] = (n < 8) ? (_Float16)W4[k * 8 + n] : (_Float16)0.f;
  }
  __syncthreads();

  // A fragment: edge_attr tile [16 x 32] from LDS (vectorized f32 runs)
  const float* er = eas + (wave * 16 + row) * 32;
  v16h a = afrag_f32(er, 0, hi);

  // three 32->64 GEMMs, 4 n-tiles each, activations to stage
#pragma unroll
  for (int n0 = 0; n0 < 64; n0 += 16) {
    int nb = (n0 + row) * 40 + 16 * hi;  // contiguous 16 halves
    v16h b1 = cat8(ldv8h(w1sT + nb), ldv8h(w1sT + nb + 8));
    v16h b2 = cat8(ldv8h(w2sT + nb), ldv8h(w2sT + nb + 8));
    v16h b3 = cat8(ldv8h(w3sT + nb), ldv8h(w3sT + nb + 8));
    v8f c1 = {}, c2 = {}, c3 = {};
    c1 = wmma16(a, b1, c1);
    c2 = wmma16(a, b2, c2);
    c3 = wmma16(a, b3, c3);
#pragma unroll
    for (int r = 0; r < 8; ++r) {
      int m = hi ? r + 8 : r;  // C: VGPR r holds row r (lanes<16) / r+8
      stage[wave][m * 128 + n0 + row] = (_Float16)fmaxf(c1[r], 0.f);
      stage[wave][m * 128 + 64 + n0 + row] =
          (_Float16)(fast_tanh(c2[r]) * fast_tanh(c3[r]));
    }
  }

  // final [16x128] @ [128x16pad] : 4 K-chunks of 32
  v8f c4 = {};
#pragma unroll
  for (int kb = 0; kb < 128; kb += 32) {
    const _Float16* sr = &stage[wave][row * 128 + kb + hi * 16];
    v16h aa = cat8(ldv8h(sr), ldv8h(sr + 8));  // runs at kb+hi*16, kb+hi*16+8
    const _Float16* wr = w4sT + row * 136 + kb + 16 * hi;
    v16h bb = cat8(ldv8h(wr), ldv8h(wr + 8));
    c4 = wmma16(aa, bb, c4);
  }
#pragma unroll
  for (int r = 0; r < 8; ++r) {
    int m = hi ? r + 8 : r;
    if (row < 8 && (ebase + m) < E)
      e_out[(ebase + m) * 8 + row] = fmaxf(c4[r], 0.f);
  }
}

// ---------------------------------------------------------------------------
// Kernel 2: Y[n,i,:] = x[n] @ Wk[i]   (Y: [N,8,128])
// Block = 256 thr covers 128 nodes (8 waves x 16-node tiles); loop over the
// 8 supports with Wk[i] staged TRANSPOSED in LDS; A fragments from global.
// ---------------------------------------------------------------------------
__global__ __launch_bounds__(256) void spectral_y_kernel(
    const float* __restrict__ x, const float* __restrict__ Wk,
    float* __restrict__ Y, int N) {
  __shared__ __attribute__((aligned(32))) _Float16 wksT[128 * 136];  // [n][k]
  int tid = threadIdx.x, wave = tid >> 5, lane = tid & 31;
  int row = lane & 15, hi = lane >> 4;
  long long nbase = ((long long)blockIdx.x * 8 + wave) * 16;

  long long nr = nbase + row;
  if (nr >= N) nr = N - 1;
  const float* xr = x + nr * 128;
  v16h a[4];
#pragma unroll
  for (int c = 0; c < 4; ++c) a[c] = afrag_f32(xr, c * 32, hi);

  for (int s = 0; s < 8; ++s) {
    __syncthreads();  // all waves done with previous wksT
    for (int i = tid; i < 16384; i += 256) {
      int k = i >> 7, n = i & 127;  // Wk[s] is [128][128], read coalesced
      wksT[n * 136 + k] = (_Float16)Wk[(long long)s * 16384 + i];
    }
    __syncthreads();

    for (int n0 = 0; n0 < 128; n0 += 16) {
      v8f acc = {};
#pragma unroll
      for (int c = 0; c < 4; ++c) {
        const _Float16* wr = wksT + (n0 + row) * 136 + c * 32 + 16 * hi;
        v16h b = cat8(ldv8h(wr), ldv8h(wr + 8));
        acc = wmma16(a[c], b, acc);
      }
#pragma unroll
      for (int r = 0; r < 8; ++r) {
        int m = hi ? r + 8 : r;
        long long n = nbase + m;
        if (n < N) Y[((n * 8 + s) << 7) + n0 + row] = acc[r];
      }
    }
  }
}

// ---------------------------------------------------------------------------
// Kernel 3: scatter  accum[dst,:] += sum_i e[edge,i] * Y[src,i,:]
// One wave per edge; lane handles 4 channels. 128 atomics/edge into the
// L2-resident 25.6MB accumulator; lookahead prefetch of gathered Y rows.
// ---------------------------------------------------------------------------
__global__ __launch_bounds__(256) void scatter_kernel(
    const long long* __restrict__ eidx, const float* __restrict__ e,
    const float* __restrict__ Y, float* __restrict__ accum, int E) {
  long long gid = (long long)blockIdx.x * 256 + threadIdx.x;
  int edge = (int)(gid >> 5);
  int lane = (int)(gid & 31);
  if (edge >= E) return;

  long long src = eidx[edge];
  long long dst = eidx[(long long)E + edge];

  int next = edge + 64;
  if (next < E) {
    long long s2 = eidx[next];
    const char* p = (const char*)(Y + (s2 << 10));
    __builtin_prefetch(p + (size_t)lane * 128, 0, 1);  // 32 lines = 4KB rows
  }

  const float* er = e + (long long)edge * 8;
  const float4* Ys = (const float4*)(Y + (src << 10));  // 8 rows x 32 float4
  float4 acc = make_float4(0.f, 0.f, 0.f, 0.f);
#pragma unroll
  for (int i = 0; i < 8; ++i) {
    float w = er[i];
    float4 v = Ys[i * 32 + lane];
    acc.x += w * v.x;
    acc.y += w * v.y;
    acc.z += w * v.z;
    acc.w += w * v.w;
  }
  float* o = accum + dst * 128 + lane * 4;
  atomicAdd(o + 0, acc.x);
  atomicAdd(o + 1, acc.y);
  atomicAdd(o + 2, acc.z);
  atomicAdd(o + 3, acc.w);
}

// ---------------------------------------------------------------------------
// Kernel 4: finalize
//   out[:, :128]   = relu(accum + bk)
//   out[:,128:192] = tanh(x@W11+b11) * tanh(x@W12+b12)
// ---------------------------------------------------------------------------
__global__ __launch_bounds__(256) void finalize_kernel(
    const float* __restrict__ x, const float* __restrict__ accum,
    const float* __restrict__ bk,
    const float* __restrict__ W11, const float* __restrict__ b11,
    const float* __restrict__ W12, const float* __restrict__ b12,
    float* __restrict__ out, int N) {
  __shared__ __attribute__((aligned(32))) _Float16 w11sT[64 * 136];  // [n][k]
  __shared__ __attribute__((aligned(32))) _Float16 w12sT[64 * 136];
  int tid = threadIdx.x, wave = tid >> 5, lane = tid & 31;
  int row = lane & 15, hi = lane >> 4;
  for (int i = tid; i < 8192; i += 256) {
    int k = i >> 6, n = i & 63;  // W [128][64] read coalesced
    w11sT[n * 136 + k] = (_Float16)W11[i];
    w12sT[n * 136 + k] = (_Float16)W12[i];
  }
  __syncthreads();

  long long nbase = ((long long)blockIdx.x * 8 + wave) * 16;
  long long nr = nbase + row;
  if (nr >= N) nr = N - 1;
  const float* xr = x + nr * 128;
  v16h a[4];
#pragma unroll
  for (int c = 0; c < 4; ++c) a[c] = afrag_f32(xr, c * 32, hi);

#pragma unroll
  for (int n0 = 0; n0 < 64; n0 += 16) {
    v8f c1 = {}, c2 = {};
#pragma unroll
    for (int c = 0; c < 4; ++c) {
      const _Float16* p1 = w11sT + (n0 + row) * 136 + c * 32 + 16 * hi;
      const _Float16* p2 = w12sT + (n0 + row) * 136 + c * 32 + 16 * hi;
      v16h b1 = cat8(ldv8h(p1), ldv8h(p1 + 8));
      v16h b2 = cat8(ldv8h(p2), ldv8h(p2 + 8));
      c1 = wmma16(a[c], b1, c1);
      c2 = wmma16(a[c], b2, c2);
    }
    float bb1 = b11[n0 + row], bb2 = b12[n0 + row];
#pragma unroll
    for (int r = 0; r < 8; ++r) {
      int m = hi ? r + 8 : r;
      long long n = nbase + m;
      if (n < N)
        out[n * 192 + 128 + n0 + row] =
            fast_tanh(c1[r] + bb1) * fast_tanh(c2[r] + bb2);
    }
  }

  for (int r = 0; r < 16; ++r) {
    long long n = nbase + r;
    if (n < N) {
      for (int cc = lane; cc < 128; cc += 32)
        out[n * 192 + cc] = fmaxf(accum[n * 128 + cc] + bk[cc], 0.f);
    }
  }
}

// ---------------------------------------------------------------------------
extern "C" void kernel_launch(void* const* d_in, const int* in_sizes, int n_in,
                              void* d_out, int out_size, void* d_ws,
                              size_t ws_size, hipStream_t stream) {
  const float* x        = (const float*)d_in[0];
  const long long* eidx = (const long long*)d_in[1];  // int64 [2,E]
  const float* eattr    = (const float*)d_in[2];
  const float* W1  = (const float*)d_in[3];
  const float* W2  = (const float*)d_in[4];
  const float* W3  = (const float*)d_in[5];
  const float* W4  = (const float*)d_in[6];
  const float* Wk  = (const float*)d_in[7];
  const float* bk  = (const float*)d_in[8];
  const float* W11 = (const float*)d_in[9];
  const float* b11 = (const float*)d_in[10];
  const float* W12 = (const float*)d_in[11];
  const float* b12 = (const float*)d_in[12];
  float* out = (float*)d_out;

  const int N = in_sizes[0] / 128;  // nodes
  const int E = in_sizes[1] / 2;    // edges

  char* ws = (char*)d_ws;
  size_t ybytes = (size_t)N * 8 * 128 * sizeof(float);  // Y [N,8,128]
  size_t ebytes = (size_t)E * 8 * sizeof(float);        // e [E,8]
  float* Y     = (float*)ws;
  float* efeat = (float*)(ws + ybytes);
  float* accum = (float*)(ws + ybytes + ebytes);        // [N,128]

  long long n4 = (long long)N * 128 / 4;
  zero_kernel<<<(unsigned)((n4 + 255) / 256), 256, 0, stream>>>((float4*)accum,
                                                                n4);
  edge_mlp_kernel<<<(unsigned)(((long long)E + 63) / 64), 128, 0, stream>>>(
      eattr, W1, W2, W3, W4, efeat, E);
  spectral_y_kernel<<<(unsigned)((N + 127) / 128), 256, 0, stream>>>(x, Wk, Y,
                                                                     N);
  scatter_kernel<<<(unsigned)(((long long)E + 7) / 8), 256, 0, stream>>>(
      eidx, efeat, Y, accum, E);
  finalize_kernel<<<(unsigned)((N + 127) / 128), 256, 0, stream>>>(
      x, accum, bk, W11, b11, W12, b12, out, N);
}
